// PairedLOHCGNN_47528108098385
// MI455X (gfx1250) — matrix operand order repaired and checked by reading, not compiled, not run on hardware
//
#include <hip/hip_runtime.h>
#include <math.h>

typedef __attribute__((ext_vector_type(16))) __bf16 v16bf;
typedef __attribute__((ext_vector_type(8)))  float  v8f;

#define NN 50000
#define EE 200000
#define GG 256
#define HH 4
#define CHN 128

__device__ __forceinline__ void atomicMaxF(float* addr, float v) {
  if (v >= 0.0f) atomicMax((int*)addr, __float_as_int(v));
  else           atomicMin((unsigned int*)addr, __float_as_uint(v));
}

__device__ __forceinline__ void edge_sd(const int* __restrict__ ei, int E, int e, int& s, int& d) {
  if (e < E) { s = ei[e]; d = ei[E + e]; } else { s = e - E; d = s; }
}

// ---------------- fp32 W [K,Fout] -> bf16 W^T [Fout,K] ----------------
__global__ void k_convw(const float* __restrict__ W, __bf16* __restrict__ Wt, int K, int Fout) {
  int t = blockIdx.x * blockDim.x + threadIdx.x;
  if (t >= K * Fout) return;
  int k = t / Fout, j = t - k * Fout;
  Wt[(size_t)j * K + k] = (__bf16)W[t];
}

// ---------------- fill ----------------
__global__ void k_fill(float* __restrict__ p, float v, long n) {
  long t = blockIdx.x * (long)blockDim.x + threadIdx.x;
  if (t < n) p[t] = v;
}

// ---------------- WMMA GEMM: C[nrows,Fout] = act(A)[nrows,K] @ Wt^T ----------------
// Each wave computes a 16x64 output tile: A fragment (with fused bias+ELU+bf16
// convert when act==1) is built once per K-step and reused by 4 independent WMMAs.
// A fp32 row-major; Wt bf16 [Fout,K] row-major. Fout must be a multiple of 64.
__global__ void k_gemm_wmma(const float* __restrict__ A, const float* __restrict__ bias, int act,
                            const __bf16* __restrict__ Bt, float* __restrict__ C,
                            int nrows, int K, int Fout) {
  const int gtid = blockIdx.x * blockDim.x + threadIdx.x;
  const int wave = gtid >> 5;
  const int lane = threadIdx.x & 31;
  const int ntg = Fout >> 6;                 // groups of 64 output columns
  const int mt  = (nrows + 15) >> 4;
  if (wave >= mt * ntg) return;
  const int tm = wave / ntg;
  const int tg = wave - tm * ntg;

  const int half = lane >> 4;                // lane 0-15 vs 16-31
  const int l15  = lane & 15;

  int m = tm * 16 + l15;
  if (m >= nrows) m = nrows - 1;             // clamp (stores are guarded)
  const float*  arow = A  + (size_t)m * K;
  const int     kA   = half * 16;            // A K-offset: contiguous k = 16*half + s (matches B)
  const __bf16* brow = Bt + (size_t)(tg * 64 + l15) * K + half * 16;
  const size_t  bstep = (size_t)16 * K;      // one 16-col sub-tile of B

  v8f acc0 = {}, acc1 = {}, acc2 = {}, acc3 = {};
  for (int k0 = 0; k0 < K; k0 += 32) {
    float av[16];
    {
      const float4* p0 = (const float4*)(arow + k0 + kA);
      float4 q0 = p0[0], q1 = p0[1];
      const float4* p1 = (const float4*)(arow + k0 + 8 + kA);
      float4 q2 = p1[0], q3 = p1[1];
      av[0]=q0.x; av[1]=q0.y; av[2]=q0.z;  av[3]=q0.w;
      av[4]=q1.x; av[5]=q1.y; av[6]=q1.z;  av[7]=q1.w;
      av[8]=q2.x; av[9]=q2.y; av[10]=q2.z; av[11]=q2.w;
      av[12]=q3.x; av[13]=q3.y; av[14]=q3.z; av[15]=q3.w;
    }
    v16bf afrag;
    if (act) {
#pragma unroll
      for (int i = 0; i < 8; ++i) {
        float x0 = av[i]     + bias[k0 + kA + i];
        float x1 = av[8 + i] + bias[k0 + 8 + kA + i];
        x0 = x0 > 0.0f ? x0 : __expf(x0) - 1.0f;   // ELU
        x1 = x1 > 0.0f ? x1 : __expf(x1) - 1.0f;
        afrag[i]     = (__bf16)x0;
        afrag[8 + i] = (__bf16)x1;
      }
    } else {
#pragma unroll
      for (int i = 0; i < 16; ++i) afrag[i] = (__bf16)av[i];
    }
    v16bf b0 = *(const v16bf*)(brow + k0);             // 32B contiguous, 32B aligned
    v16bf b1v = *(const v16bf*)(brow + bstep + k0);
    v16bf b2v = *(const v16bf*)(brow + 2 * bstep + k0);
    v16bf b3v = *(const v16bf*)(brow + 3 * bstep + k0);
    acc0 = __builtin_amdgcn_wmma_f32_16x16x32_bf16(false, afrag, false, b0,  (short)0, acc0, false, false);
    acc1 = __builtin_amdgcn_wmma_f32_16x16x32_bf16(false, afrag, false, b1v, (short)0, acc1, false, false);
    acc2 = __builtin_amdgcn_wmma_f32_16x16x32_bf16(false, afrag, false, b2v, (short)0, acc2, false, false);
    acc3 = __builtin_amdgcn_wmma_f32_16x16x32_bf16(false, afrag, false, b3v, (short)0, acc3, false, false);
  }
  const int mbase = tm * 16 + half * 8;
  const int ncb   = tg * 64 + l15;
#pragma unroll
  for (int r = 0; r < 8; ++r) {
    int mm = mbase + r;
    if (mm < nrows) {
      float* crow = C + (size_t)mm * Fout + ncb;
      crow[0]  = acc0[r];
      crow[16] = acc1[r];
      crow[32] = acc2[r];
      crow[48] = acc3[r];
    }
  }
}

// ---------------- per-(node,head) attention scores: one wave per row ----------------
__global__ void k_scores(const float* __restrict__ Hm, const float* __restrict__ a_s,
                         const float* __restrict__ a_d, float* __restrict__ ssrc,
                         float* __restrict__ sdst, int n_rows, int Hn, int Cn) {
  int wid  = (blockIdx.x * blockDim.x + threadIdx.x) >> 5;
  int lane = threadIdx.x & 31;
  if (wid >= n_rows * Hn) return;
  int n = wid / Hn, h = wid - n * Hn;
  const float* row = Hm + (size_t)n * (Hn * Cn) + h * Cn;
  const float* av  = a_s + h * Cn;
  const float* bv  = a_d + h * Cn;
  float s1 = 0.0f, s2 = 0.0f;
  for (int c = lane; c < Cn; c += 32) { float x = row[c]; s1 += x * av[c]; s2 += x * bv[c]; }
  for (int off = 16; off; off >>= 1) { s1 += __shfl_down(s1, off, 32); s2 += __shfl_down(s2, off, 32); }
  if (lane == 0) { ssrc[wid] = s1; sdst[wid] = s2; }
}

// ---------------- edge pass 1: segment max over dst ----------------
__global__ void k_edge_max(const int* __restrict__ ei, int E, int Etot,
                           const float* __restrict__ ssrc, const float* __restrict__ sdst,
                           float* __restrict__ emax, int Hn) {
  int t = blockIdx.x * blockDim.x + threadIdx.x;
  if (t >= Etot * Hn) return;
  int e = t / Hn, h = t - e * Hn;
  int s, d; edge_sd(ei, E, e, s, d);
  float v = ssrc[s * Hn + h] + sdst[d * Hn + h];
  v = v > 0.0f ? v : 0.2f * v;                     // leaky_relu
  atomicMaxF(emax + d * Hn + h, v);
}

// ---------------- edge pass 2: exp + segment sum (store p) ----------------
__global__ void k_edge_exp(const int* __restrict__ ei, int E, int Etot,
                           const float* __restrict__ ssrc, const float* __restrict__ sdst,
                           const float* __restrict__ emax, float* __restrict__ denom,
                           float* __restrict__ pbuf, int Hn) {
  int t = blockIdx.x * blockDim.x + threadIdx.x;
  if (t >= Etot * Hn) return;
  int e = t / Hn, h = t - e * Hn;
  int s, d; edge_sd(ei, E, e, s, d);
  float v = ssrc[s * Hn + h] + sdst[d * Hn + h];
  v = v > 0.0f ? v : 0.2f * v;
  float m = emax[d * Hn + h];
  if (!__builtin_isfinite(m)) m = 0.0f;
  float pv = __expf(v - m);
  pbuf[t] = pv;
  atomicAdd(denom + d * Hn + h, pv);
}

// ---------------- edge pass 3: scatter h[src]*alpha into out[dst]; wave per (e,h) ----------------
__global__ void k_edge_scatter(const int* __restrict__ ei, int E, int Etot,
                               const float* __restrict__ Hm, const float* __restrict__ pbuf,
                               const float* __restrict__ denom, float* __restrict__ OUT,
                               int Hn, int Cn) {
  int wid  = (blockIdx.x * blockDim.x + threadIdx.x) >> 5;
  int lane = threadIdx.x & 31;
  if (wid >= Etot * Hn) return;
  int e = wid / Hn, h = wid - e * Hn;
  int s, d; edge_sd(ei, E, e, s, d);
  float alpha = pbuf[wid] / (denom[d * Hn + h] + 1e-16f);
  size_t bs = (size_t)s * (Hn * Cn) + h * Cn + lane * 4;
  size_t bd = (size_t)d * (Hn * Cn) + h * Cn + lane * 4;
  float4 hv = *(const float4*)(Hm + bs);
  atomicAdd(OUT + bd + 0, hv.x * alpha);
  atomicAdd(OUT + bd + 1, hv.y * alpha);
  atomicAdd(OUT + bd + 2, hv.z * alpha);
  atomicAdd(OUT + bd + 3, hv.w * alpha);
}

// ---------------- global max pool (+bias) over batch ----------------
__global__ void k_pool(const float* __restrict__ O3, const float* __restrict__ b3,
                       const int* __restrict__ batch, float* __restrict__ emb, int n_rows, int Cn) {
  int t = blockIdx.x * blockDim.x + threadIdx.x;
  if (t >= n_rows * Cn) return;
  int n = t / Cn, c = t - n * Cn;
  atomicMaxF(emb + batch[n] * Cn + c, O3[t] + b3[c]);
}

// ---------------- MLP head: one block per molecule ----------------
__global__ void k_mlp(const float* __restrict__ e1, const float* __restrict__ e2,
                      const float* __restrict__ mw1, const float* __restrict__ mb1,
                      const float* __restrict__ mw2, const float* __restrict__ mb2,
                      float* __restrict__ out) {
  int g = blockIdx.x, j = threadIdx.x;   // 128 threads
  __shared__ float sh[128];
  float acc = mb1[j];
  for (int k = 0; k < 128; ++k) {
    float z = e1[g * 128 + k]; if (!__builtin_isfinite(z)) z = 0.0f;
    acc += z * mw1[k * 128 + j];
  }
  for (int k = 0; k < 128; ++k) {
    float z = e2[g * 128 + k]; if (!__builtin_isfinite(z)) z = 0.0f;
    acc += z * mw1[(128 + k) * 128 + j];
  }
  acc = fmaxf(acc, 0.0f);
  sh[j] = acc * mw2[j];
  __syncthreads();
  for (int s = 64; s; s >>= 1) { if (j < s) sh[j] += sh[j + s]; __syncthreads(); }
  if (j == 0) out[g] = sh[0] + mb2[0];
}

extern "C" void kernel_launch(void* const* d_in, const int* in_sizes, int n_in,
                              void* d_out, int out_size, void* d_ws, size_t ws_size,
                              hipStream_t stream) {
  (void)in_sizes; (void)n_in; (void)out_size; (void)ws_size;
  const int N = NN, E = EE, G = GG, H = HH, CH = CHN, D = H * CH;
  const int Etot = E + N;

  const float* x1  = (const float*)d_in[0];
  const int*   ei1 = (const int*)  d_in[1];
  const int*   bt1 = (const int*)  d_in[2];
  const float* x2  = (const float*)d_in[3];
  const int*   ei2 = (const int*)  d_in[4];
  const int*   bt2 = (const int*)  d_in[5];
  const float* W1  = (const float*)d_in[6];
  const float* as1 = (const float*)d_in[7];
  const float* ad1 = (const float*)d_in[8];
  const float* b1  = (const float*)d_in[9];
  const float* W2  = (const float*)d_in[10];
  const float* as2 = (const float*)d_in[11];
  const float* ad2 = (const float*)d_in[12];
  const float* b2  = (const float*)d_in[13];
  const float* W3  = (const float*)d_in[14];
  const float* as3 = (const float*)d_in[15];
  const float* ad3 = (const float*)d_in[16];
  const float* b3  = (const float*)d_in[17];
  const float* mw1 = (const float*)d_in[18];
  const float* mb1 = (const float*)d_in[19];
  const float* mw2 = (const float*)d_in[20];
  const float* mb2 = (const float*)d_in[21];

  char* ws = (char*)d_ws;
  auto carve = [&](size_t bytes) { void* p = (void*)ws; ws += (bytes + 255) & ~(size_t)255; return p; };
  __bf16* wt1  = (__bf16*)carve((size_t)D * 64 * 2);       // W1^T [512,64]
  __bf16* wt2  = (__bf16*)carve((size_t)D * D * 2);        // W2^T [512,512]
  __bf16* wt3  = (__bf16*)carve((size_t)CH * D * 2);       // W3^T [128,512]
  float* Hbuf  = (float*)carve((size_t)N * D * 4);         // h = x@W
  float* Obuf  = (float*)carve((size_t)N * D * 4);         // aggregated out
  float* ssrc  = (float*)carve((size_t)N * H * 4);
  float* sdst  = (float*)carve((size_t)N * H * 4);
  float* emaxb = (float*)carve((size_t)N * H * 4);
  float* denb  = (float*)carve((size_t)N * H * 4);
  float* pbuf  = (float*)carve((size_t)Etot * H * 4);
  float* emb1  = (float*)carve((size_t)G * CH * 4);
  float* emb2  = (float*)carve((size_t)G * CH * 4);

  auto cdiv = [](long a, long b) { return (a + b - 1) / b; };

  // weight transpose + bf16 convert (small, done each call — deterministic)
  k_convw<<<(unsigned)cdiv((long)64 * D, 256), 256, 0, stream>>>(W1, wt1, 64, D);
  k_convw<<<(unsigned)cdiv((long)D * D, 256), 256, 0, stream>>>(W2, wt2, D, D);
  k_convw<<<(unsigned)cdiv((long)D * CH, 256), 256, 0, stream>>>(W3, wt3, D, CH);

  auto run_layer = [&](const float* A, const float* biasPrev, int act, const __bf16* Wt,
                       int K, int Fout, int Hn, const float* a_s, const float* a_d,
                       const int* ei) {
    long waves = cdiv(N, 16) * (Fout / 64);  // 16x64 tile per wave
    k_gemm_wmma<<<(unsigned)cdiv(waves * 32, 256), 256, 0, stream>>>(A, biasPrev, act, Wt, Hbuf, N, K, Fout);
    long srow = (long)N * Hn;
    k_scores<<<(unsigned)cdiv(srow * 32, 256), 256, 0, stream>>>(Hbuf, a_s, a_d, ssrc, sdst, N, Hn, CHN);
    k_fill<<<(unsigned)cdiv(srow, 256), 256, 0, stream>>>(emaxb, -INFINITY, srow);
    k_fill<<<(unsigned)cdiv(srow, 256), 256, 0, stream>>>(denb, 0.0f, srow);
    k_fill<<<(unsigned)cdiv((long)N * Fout, 256), 256, 0, stream>>>(Obuf, 0.0f, (long)N * Fout);
    long et = (long)Etot * Hn;
    k_edge_max<<<(unsigned)cdiv(et, 256), 256, 0, stream>>>(ei, E, Etot, ssrc, sdst, emaxb, Hn);
    k_edge_exp<<<(unsigned)cdiv(et, 256), 256, 0, stream>>>(ei, E, Etot, ssrc, sdst, emaxb, denb, pbuf, Hn);
    k_edge_scatter<<<(unsigned)cdiv(et * 32, 256), 256, 0, stream>>>(ei, E, Etot, Hbuf, pbuf, denb, Obuf, Hn, CHN);
  };

  auto run_encoder = [&](const float* x, const int* ei, const int* batch, float* emb) {
    run_layer(x,    nullptr, 0, wt1, 64, D,  H, as1, ad1, ei);  // layer1 (raw x)
    run_layer(Obuf, b1,      1, wt2, D,  D,  H, as2, ad2, ei);  // layer2, elu(.+b1) fused
    run_layer(Obuf, b2,      1, wt3, D,  CH, 1, as3, ad3, ei);  // layer3, elu(.+b2) fused
    k_fill<<<(unsigned)cdiv((long)G * CH, 256), 256, 0, stream>>>(emb, -INFINITY, (long)G * CH);
    k_pool<<<(unsigned)cdiv((long)N * CH, 256), 256, 0, stream>>>(Obuf, b3, batch, emb, N, CH);
  };

  run_encoder(x1, ei1, bt1, emb1);
  run_encoder(x2, ei2, bt2, emb2);
  k_mlp<<<G, 128, 0, stream>>>(emb1, emb2, mw1, mb1, mw2, mb2, (float*)d_out);
}
